// SparseMHA_83691732730622
// MI455X (gfx1250) — matrix operand order repaired and checked by reading, not compiled, not run on hardware
//
#include <hip/hip_runtime.h>
#include <hip/hip_bf16.h>

// SparseMHA on gfx1250 (MI455X): bf16-resident WMMA pipeline + bisection sparsemax.
// B=1, S=2048, E=1024, H=16, D=64.
// GEMMs stage the shared A panel in LDS via the gfx1250 async global->LDS path.

#define SQ  2048
#define EMB 1024
#define NH  16
#define HD  64

typedef __attribute__((ext_vector_type(16))) __bf16 v16bf;
typedef __attribute__((ext_vector_type(8)))  __bf16 v8bf;
typedef __attribute__((ext_vector_type(8)))  float  v8f;
typedef __attribute__((ext_vector_type(4)))  int    v4i;

#define AS_LDS __attribute__((address_space(3)))

// ---------------------------------------------------------------------------
// Async global->LDS 16-byte copy (ASYNCcnt path); falls back to load+ds_store.
// Probe result: builtin exists; param0 is a generic v4i* (flat source address).
// ---------------------------------------------------------------------------
static __device__ __forceinline__ void async_copy_b128(const void* g, void* l) {
#if defined(__has_builtin) && __has_builtin(__builtin_amdgcn_global_load_async_to_lds_b128)
  __builtin_amdgcn_global_load_async_to_lds_b128((v4i*)g, (AS_LDS v4i*)l, 0, 0);
#else
  *reinterpret_cast<uint4*>(l) = *reinterpret_cast<const uint4*>(g);
#endif
}

static __device__ __forceinline__ void wait_async_done() {
  // ASYNCcnt == 0; harmless no-op if the fallback (sync) copy path was used.
  asm volatile("s_wait_asynccnt 0" ::: "memory");
}

// ---------------------------------------------------------------------------
// Fragment loaders (wave32 WMMA 16x16x32 bf16 layouts):
//   A (16x32): lane = 16h + r holds row r; element i -> K = i + 16h
//              == one contiguous 16-element run at +16h.
//   B (32x16): lane = 16h + r holds column r; element i -> K = i + 16h
//              == one contiguous 16-element run at +16h.
// ---------------------------------------------------------------------------
static __device__ __forceinline__ v16bf load_a_bf16(const __bf16* __restrict__ p) {
  v8bf lo = *reinterpret_cast<const v8bf*>(p);        // 16B
  v8bf hi = *reinterpret_cast<const v8bf*>(p + 8);    // 16B
  return __builtin_shufflevector(lo, hi, 0,1,2,3,4,5,6,7,8,9,10,11,12,13,14,15);
}

static __device__ __forceinline__ v16bf load_b_bf16(const __bf16* __restrict__ p) {
  return *reinterpret_cast<const v16bf*>(p);          // 32B
}

// ---------------------------------------------------------------------------
// Elementwise f32 -> bf16 (RNE), 8 elements per thread. n must be 8-aligned.
// ---------------------------------------------------------------------------
__global__ void cvt_f32_bf16(const float* __restrict__ src, __bf16* __restrict__ dst, int n) {
  const int i = (blockIdx.x * blockDim.x + threadIdx.x) * 8;
  if (i >= n) return;
  const float4 a = *reinterpret_cast<const float4*>(src + i);
  const float4 b = *reinterpret_cast<const float4*>(src + i + 4);
  v8bf o;
  o[0] = (__bf16)a.x; o[1] = (__bf16)a.y; o[2] = (__bf16)a.z; o[3] = (__bf16)a.w;
  o[4] = (__bf16)b.x; o[5] = (__bf16)b.y; o[6] = (__bf16)b.z; o[7] = (__bf16)b.w;
  *reinterpret_cast<v8bf*>(dst + i) = o;
}

// ---------------------------------------------------------------------------
// C = A @ W^T + bias.  A: Mx1024 bf16 row-major, W: Nx1024 bf16 row-major.
// MODE 0: C f32 MxN row-major (final output)
// MODE 1: C bf16 MxN row-major (Q/K projections)
// MODE 2: C bf16 transposed NxM row-major (V projection -> V^T, packed stores)
// Block = 8 waves; the block's 16-row A panel (32 KB, contiguous since K=1024
// is the full row) is async-staged into LDS once and shared by all waves.
// ---------------------------------------------------------------------------
template <int MODE>
__global__ void gemm_bf16(const __bf16* __restrict__ A,
                          const __bf16* __restrict__ W,
                          const float*  __restrict__ bias,
                          void* __restrict__ Cout,
                          int M, int N) {
  constexpr int K = EMB;
  __shared__ __bf16 sA[16 * K];                      // 32 KB

  const int lane = threadIdx.x & 31;
  const int wave = threadIdx.x >> 5;
  const int h    = lane >> 4;
  const int r    = lane & 15;
  const int n0   = blockIdx.x * 128 + wave * 16;
  const int m0   = blockIdx.y * 16;

  // ---- async-stage A panel: flat 32 KB copy, 128 B per thread ----
  {
    const char* g = reinterpret_cast<const char*>(A + (size_t)m0 * K) + threadIdx.x * 128;
    char*       l = reinterpret_cast<char*>(sA) + threadIdx.x * 128;
#pragma unroll
    for (int c = 0; c < 8; ++c) async_copy_b128(g + c * 16, l + c * 16);
    wait_async_done();
  }
  __syncthreads();

  const __bf16* arow = sA + r * K;                   // LDS-resident A row
  const __bf16* brow = W + (size_t)(n0 + r) * K;

  v8f c = {};
  for (int k0 = 0; k0 < K; k0 += 32) {
    v16bf a = load_a_bf16(arow + k0 + 16 * h);       // ds_load_b128 x2
    v16bf b = load_b_bf16(brow + k0 + 16 * h);       // global 32B
    c = __builtin_amdgcn_wmma_f32_16x16x32_bf16(false, a, false, b, (short)0, c, false, false);
    if (k0 + 256 < K) __builtin_prefetch(brow + k0 + 256, 0, 1);  // global_prefetch_b8
  }
  const float bn = bias[n0 + r];

  if constexpr (MODE == 0) {
    float* C = (float*)Cout;
#pragma unroll
    for (int j = 0; j < 8; ++j)
      C[(size_t)(m0 + j + 8 * h) * N + (n0 + r)] = c[j] + bn;
  } else if constexpr (MODE == 1) {
    __bf16* C = (__bf16*)Cout;
#pragma unroll
    for (int j = 0; j < 8; ++j)
      C[(size_t)(m0 + j + 8 * h) * N + (n0 + r)] = (__bf16)(c[j] + bn);
  } else {
    // Transposed: row (n0+r), cols m0+8h .. m0+8h+7 -> one packed 16B store.
    __bf16* C = (__bf16*)Cout;
    v8bf o;
#pragma unroll
    for (int j = 0; j < 8; ++j) o[j] = (__bf16)(c[j] + bn);
    *reinterpret_cast<v8bf*>(C + (size_t)(n0 + r) * M + m0 + 8 * h) = o;
  }
}

// ---------------------------------------------------------------------------
// Fused sparsemax attention for one (head, 16-query-row) tile.
// Block = 256 threads (8 waves).
// Dynamic LDS: 16 x 2048 f32 scores (128 KB) + 16 x 2048 bf16 probs (64 KB).
//   Stage 1: scores = (Q K^T)/8 with key mask  -> f32 LDS
//   Stage 2: sparsemax per row via bisection on tau; probs -> bf16 LDS
//   Stage 3: attended = P @ V_head (V^T layout) -> bf16 ATT[S][E]
// ---------------------------------------------------------------------------
__global__ void sparsemax_attn(const __bf16* __restrict__ Qb,   // S x E bf16
                               const __bf16* __restrict__ Kb,   // S x E bf16
                               const __bf16* __restrict__ Vt,   // E x S bf16 (transposed)
                               const int*    __restrict__ mask, // S
                               __bf16* __restrict__ ATT) {      // S x E bf16
  extern __shared__ char smem[];
  float*  sS  = reinterpret_cast<float*>(smem);                               // [16][SQ] f32
  __bf16* sPb = reinterpret_cast<__bf16*>(smem + 16 * SQ * sizeof(float));    // [16][SQ] bf16

  const int lane = threadIdx.x & 31;
  const int wave = threadIdx.x >> 5;
  const int h    = lane >> 4;
  const int r    = lane & 15;
  const int q0   = blockIdx.x * 16;
  const int head = blockIdx.y;

  const __bf16* Qh = Qb + head * HD;
  const __bf16* Kh = Kb + head * HD;

  // ---- Stage 1: score tiles across the key dimension (256 keys per wave) ----
  v16bf aq0 = load_a_bf16(Qh + (size_t)(q0 + r) * EMB + 0  + 16 * h);  // d = 0..31
  v16bf aq1 = load_a_bf16(Qh + (size_t)(q0 + r) * EMB + 32 + 16 * h);  // d = 32..63

  for (int kt = wave * 256; kt < wave * 256 + 256; kt += 16) {
    v8f c = {};
    v16bf b0 = load_b_bf16(Kh + (size_t)(kt + r) * EMB + 0  + 16 * h);
    c = __builtin_amdgcn_wmma_f32_16x16x32_bf16(false, aq0, false, b0, (short)0, c, false, false);
    v16bf b1 = load_b_bf16(Kh + (size_t)(kt + r) * EMB + 32 + 16 * h);
    c = __builtin_amdgcn_wmma_f32_16x16x32_bf16(false, aq1, false, b1, (short)0, c, false, false);

    const bool dead = (mask[kt + r] == 0);       // key mask, column = kt + r
#pragma unroll
    for (int j = 0; j < 8; ++j)
      sS[(j + 8 * h) * SQ + (kt + r)] = dead ? -1e9f : c[j] * 0.125f;  // 1/sqrt(64)
  }
  __syncthreads();

  // ---- Stage 2: sparsemax per row (2 rows per wave) ----
  for (int row = wave * 2; row < wave * 2 + 2; ++row) {
    float*  z  = sS  + row * SQ;
    __bf16* pb = sPb + row * SQ;

    float mx = -3.4e38f;
    for (int j = lane; j < SQ; j += 32) mx = fmaxf(mx, z[j]);
#pragma unroll
    for (int off = 16; off; off >>= 1) mx = fmaxf(mx, __shfl_xor(mx, off, 32));

    // tau solves sum(relu(z - tau)) == 1; root lies in [mx-1, mx].
    float lo = mx - 1.0f, hi = mx;
    for (int it = 0; it < 30; ++it) {
      const float mid = 0.5f * (lo + hi);
      float s = 0.0f;
      for (int j = lane; j < SQ; j += 32) s += fmaxf(z[j] - mid, 0.0f);
#pragma unroll
      for (int off = 16; off; off >>= 1) s += __shfl_xor(s, off, 32);
      if (s > 1.0f) lo = mid; else hi = mid;
    }
    const float tau = 0.5f * (lo + hi);
    for (int j = lane; j < SQ; j += 32) pb[j] = (__bf16)fmaxf(z[j] - tau, 0.0f);
  }
  __syncthreads();

  // ---- Stage 3: attended(16x64) = P(16x2048) @ V_head(2048x64), V^T layout ----
  if (wave < 4) {
    const int d0 = wave * 16;
    const __bf16* vtcol = Vt + (size_t)(head * HD + d0 + r) * SQ;  // column d0+r of V_head
    v8f c = {};
    for (int k0 = 0; k0 < SQ; k0 += 32) {
      v16bf a = load_a_bf16(sPb + r * SQ + k0 + 16 * h);     // ds_load_b128 x2
      v16bf b = load_b_bf16(vtcol + k0 + 16 * h);            // contiguous 32B
      c = __builtin_amdgcn_wmma_f32_16x16x32_bf16(false, a, false, b, (short)0, c, false, false);
    }
#pragma unroll
    for (int j = 0; j < 8; ++j)
      ATT[(size_t)(q0 + j + 8 * h) * EMB + head * HD + d0 + r] = (__bf16)c[j];
  }
}

// ---------------------------------------------------------------------------
extern "C" void kernel_launch(void* const* d_in, const int* in_sizes, int n_in,
                              void* d_out, int out_size, void* d_ws, size_t ws_size,
                              hipStream_t stream) {
  (void)in_sizes; (void)n_in; (void)out_size; (void)ws_size;

  const float* query = (const float*)d_in[0];
  const float* key   = (const float*)d_in[1];
  const float* value = (const float*)d_in[2];
  const int*   mask  = (const int*)  d_in[3];
  const float* Wq    = (const float*)d_in[4];
  const float* bq    = (const float*)d_in[5];
  const float* Wk    = (const float*)d_in[6];
  const float* bk    = (const float*)d_in[7];
  const float* Wv    = (const float*)d_in[8];
  const float* bv    = (const float*)d_in[9];
  const float* Wo    = (const float*)d_in[10];
  const float* bo    = (const float*)d_in[11];
  float* out = (float*)d_out;

  // Workspace layout (bf16): inputs (3x4MB), weights (4x2MB),
  // Qp/Kp (2x4MB), Vt (4MB), attended (4MB)  => 36 MB total.
  const size_t SE = (size_t)SQ * EMB, EE = (size_t)EMB * EMB;
  __bf16* xq = (__bf16*)d_ws;
  __bf16* xk = xq + SE;
  __bf16* xv = xk + SE;
  __bf16* wq = xv + SE;
  __bf16* wk = wq + EE;
  __bf16* wv = wk + EE;
  __bf16* wo = wv + EE;
  __bf16* Qp = wo + EE;
  __bf16* Kp = Qp + SE;
  __bf16* Vt = Kp + SE;
  __bf16* Ap = Vt + SE;

  const dim3 blk(256);

  // 1) one-shot f32 -> bf16 conversions (bandwidth-trivial)
  cvt_f32_bf16<<<dim3((int)(SE / 8 / 256)), blk, 0, stream>>>(query, xq, (int)SE);
  cvt_f32_bf16<<<dim3((int)(SE / 8 / 256)), blk, 0, stream>>>(key,   xk, (int)SE);
  cvt_f32_bf16<<<dim3((int)(SE / 8 / 256)), blk, 0, stream>>>(value, xv, (int)SE);
  cvt_f32_bf16<<<dim3((int)(EE / 8 / 256)), blk, 0, stream>>>(Wq, wq, (int)EE);
  cvt_f32_bf16<<<dim3((int)(EE / 8 / 256)), blk, 0, stream>>>(Wk, wk, (int)EE);
  cvt_f32_bf16<<<dim3((int)(EE / 8 / 256)), blk, 0, stream>>>(Wv, wv, (int)EE);
  cvt_f32_bf16<<<dim3((int)(EE / 8 / 256)), blk, 0, stream>>>(Wo, wo, (int)EE);

  // 2) projections (bf16 in, bf16 out; V written transposed)
  const dim3 ggrid(EMB / 128, SQ / 16);
  gemm_bf16<1><<<ggrid, blk, 0, stream>>>(xq, wq, bq, Qp, SQ, EMB);
  gemm_bf16<1><<<ggrid, blk, 0, stream>>>(xk, wk, bk, Kp, SQ, EMB);
  gemm_bf16<2><<<ggrid, blk, 0, stream>>>(xv, wv, bv, Vt, SQ, EMB);

  // 3) fused sparsemax attention
  const int smemBytes = 16 * SQ * (int)sizeof(float) + 16 * SQ * (int)sizeof(__bf16); // 192 KB
  (void)hipFuncSetAttribute(reinterpret_cast<const void*>(sparsemax_attn),
                            hipFuncAttributeMaxDynamicSharedMemorySize, smemBytes);
  sparsemax_attn<<<dim3(SQ / 16, NH), blk, smemBytes, stream>>>(Qp, Kp, Vt, mask, Ap);

  // 4) output projection (bf16 in, f32 out)
  gemm_bf16<0><<<ggrid, blk, 0, stream>>>(Ap, wo, bo, out, SQ, EMB);
}
